// SwinTransformerBlock_47210280517737
// MI455X (gfx1250) — matrix operand order repaired and checked by reading, not compiled, not run on hardware
//
#include <hip/hip_runtime.h>
#include <math.h>

#define DEVI __device__ __forceinline__

// ---------------------------------------------------------------- types
typedef __attribute__((ext_vector_type(16))) __bf16 v16bf;
typedef __attribute__((ext_vector_type(8)))  float  v8f;
typedef unsigned int u32x4 __attribute__((ext_vector_type(4)));
typedef int          i32x4 __attribute__((ext_vector_type(4)));
typedef int          i32x8 __attribute__((ext_vector_type(8)));

#if defined(__HIP_DEVICE_COMPILE__) && __has_builtin(__builtin_amdgcn_tensor_load_to_lds)
#define HAVE_TDM 1
#else
#define HAVE_TDM 0
#endif

// ---------------------------------------------------------------- helpers
DEVI unsigned short f32_to_bf16(float f) {
    union { float f; unsigned int u; } x; x.f = f;
    unsigned int r = x.u + 0x7fffu + ((x.u >> 16) & 1u);   // RNE
    return (unsigned short)(r >> 16);
}
DEVI float bf16_to_f32(unsigned short s) {
    union { unsigned int u; float f; } x; x.u = ((unsigned int)s) << 16;
    return x.f;
}

// window-order token d -> image token index with +SS roll (also the LN1
// gather index: dest window pos p reads shifted image (p+4)%128).
DEVI int win_to_tok(int d) {
    int win  = d >> 6, n = d & 63;
    int b    = win >> 8, widx = win & 255;
    int wh   = widx >> 4, ww = widx & 15;
    int r    = n >> 3,  c  = n & 7;
    int hh   = (wh * 8 + r + 4) & 127;
    int wp   = (ww * 8 + c + 4) & 127;
    return (b << 14) + (hh << 7) + wp;
}

// A-matrix fragment (16x32 bf16): lane<16 holds K {kb..kb+15},
// lane>=16 holds K {kb+16..kb+31}; row = base + lane%16.
DEVI v16bf load_a(const unsigned short* base, int ld, int kb, int lane) {
    int r = lane & 15, h = (lane >> 4) & 1;
    const unsigned short* p = base + (size_t)r * ld + kb + h * 16;
    union { uint4 u[2]; v16bf v; } U;
    U.u[0] = *(const uint4*)p;
    U.u[1] = *(const uint4*)(p + 8);
    return U.v;
}
// B-matrix fragment (32x16 bf16), memory is (N,K) row-major: lane%16 = column N,
// lane half selects contiguous K block of 16.
DEVI v16bf load_b(const unsigned short* base, int ld, int kb, int lane) {
    int r = lane & 15, h = (lane >> 4) & 1;
    const unsigned short* p = base + (size_t)r * ld + kb + h * 16;
    union { uint4 u[2]; v16bf v; } U;
    U.u[0] = *(const uint4*)p;
    U.u[1] = *(const uint4*)(p + 8);
    return U.v;
}
// B fragment for V^T (N=head dim, K=token): gather strided from (token,768) rows.
DEVI v16bf load_vT(const unsigned short* vbase, int n0, int kb, int lane) {
    int r = lane & 15, h = (lane >> 4) & 1;
    union { unsigned short s[16]; v16bf v; } U;
    const unsigned short* p = vbase + n0 + r;
#pragma unroll
    for (int e = 0; e < 16; ++e) U.s[e] = p[(size_t)(kb + h * 16 + e) * 768];
    return U.v;
}

DEVI v8f wmma_bf16(v16bf a, v16bf b, v8f c) {
    return __builtin_amdgcn_wmma_f32_16x16x32_bf16(false, a, false, b,
                                                   (short)0, c, false, false);
}

#if HAVE_TDM
// TDM 2-D tile load: 32 rows x 256 bf16 from global (row stride = K elems)
// into LDS with 4-DWORD padding every 128 DWORDs (row stride 528 B in LDS).
// D# layout per CDNA5 ISA.
DEVI void tdm_load_a(unsigned int lds_off, const unsigned short* gptr,
                     unsigned int k_stride) {
    unsigned long long ga = (unsigned long long)(size_t)gptr;
    u32x4 g0;
    g0[0] = 1u;                                   // count=1, user descriptor
    g0[1] = lds_off;                              // lds_addr (bytes)
    g0[2] = (unsigned int)ga;                     // global_addr[31:0]
    g0[3] = (unsigned int)((ga >> 32) & 0x01FFFFFFu) | (2u << 30);  // type=2
    const unsigned int td0 = 256u, td1 = 32u;     // tensor dims (elems)
    const unsigned int tile0 = 256u, tile1 = 32u; // tile dims
    i32x8 g1;
    // w0: wg_mask=0 | data_size(2B)=1<<16 | pad_enable<<20 |
    //     pad_interval(128DW)=6<<22 | pad_amount(4DW)=3<<25
    g1[0] = (int)((1u << 16) | (1u << 20) | (6u << 22) | (3u << 25));
    g1[1] = (int)((td0 & 0xFFFFu) << 16);         // abar_addr=0 | td0[15:0]
    g1[2] = (int)((td0 >> 16) | ((td1 & 0xFFFFu) << 16));
    g1[3] = (int)((td1 >> 16) | (tile0 << 16));   // td1[31:16] | tile_dim0
    g1[4] = (int)(tile1);                         // tile_dim1 | tile_dim2=0
    g1[5] = (int)k_stride;                        // tensor_dim0_stride[31:0]
    g1[6] = 0;                                    // stride0[47:32] | stride1[15:0]
    g1[7] = 0;                                    // stride1[47:16]
    i32x4 gz = {0, 0, 0, 0};                      // groups 2/3: 2-D tensor
#if defined(__clang_major__) && (__clang_major__ >= 23)
    i32x8 gz8 = {0, 0, 0, 0, 0, 0, 0, 0};
    __builtin_amdgcn_tensor_load_to_lds(g0, g1, gz, gz, gz8, 0);
#else
    __builtin_amdgcn_tensor_load_to_lds(g0, g1, gz, gz, 0);
#endif
}
#endif

// ---------------------------------------------------------------- weight cvt
__global__ void k_cvt(const float* __restrict__ src,
                      unsigned short* __restrict__ dst, int n) {
    int i = blockIdx.x * 256 + threadIdx.x;
    if (i < n) dst[i] = f32_to_bf16(src[i]);
}

// ---------------------------------------------------------------- layernorm
// one wave per token; SHIFT=true fuses roll(-4,-4)+window partition (gather).
template <bool SHIFT>
__global__ __launch_bounds__(128) void k_ln(const float* __restrict__ x,
                                            const float* __restrict__ g,
                                            const float* __restrict__ b,
                                            unsigned short* __restrict__ out) {
    int lane = threadIdx.x & 31, wv = threadIdx.x >> 5;
    int d   = blockIdx.x * 4 + wv;
    int src = SHIFT ? win_to_tok(d) : d;
    const float* xp = x + (size_t)src * 256 + lane * 8;
    float xv[8];
    *(float4*)(xv)     = *(const float4*)xp;
    *(float4*)(xv + 4) = *(const float4*)(xp + 4);
    float s = 0.f, q = 0.f;
#pragma unroll
    for (int e = 0; e < 8; ++e) { s += xv[e]; q += xv[e] * xv[e]; }
#pragma unroll
    for (int off = 16; off > 0; off >>= 1) {
        s += __shfl_xor(s, off, 32);
        q += __shfl_xor(q, off, 32);
    }
    float mean = s * (1.0f / 256.0f);
    float var  = q * (1.0f / 256.0f) - mean * mean;
    float inv  = rsqrtf(var + 1e-5f);
    union { unsigned short s8[8]; uint4 u; } O;
#pragma unroll
    for (int e = 0; e < 8; ++e) {
        int c = lane * 8 + e;
        O.s8[e] = f32_to_bf16((xv[e] - mean) * inv * g[c] + b[c]);
    }
    *(uint4*)(out + (size_t)d * 256 + lane * 8) = O.u;
}

// ---------------------------------------------------------------- WMMA GEMM
// Out(M,N) = A(M,K) * W(N,K)^T + bias.  Per-wave 32x64 tile, 4 waves/block.
// The 32xK A stripe is shared by all 4 waves: staged into LDS by the TDM
// (256-elem K slices, padded rows -> conflict-free ds_load_b128), weights
// stream from global/L2.
enum { EPI_BF16 = 0, EPI_GELU = 1, EPI_PROJ = 2, EPI_FC2 = 3 };

#define LDA_P 264  // padded LDS row stride in elements (528 B = 33*16 B)

template <int EPI>
__global__ __launch_bounds__(128) void k_gemm(
    const unsigned short* __restrict__ A, const unsigned short* __restrict__ W,
    const float* __restrict__ bias, void* __restrict__ out, int M, int N, int K,
    const float* __restrict__ extra, float* __restrict__ x2out) {
    __shared__ unsigned short sAT[32 * LDA_P];

    const int lane = threadIdx.x & 31;
    const int wv   = threadIdx.x >> 5;
    const int m0   = blockIdx.x * 32;
    const int n0   = (blockIdx.y * 4 + wv) * 64;

    const unsigned short* W0 = W + (size_t)n0 * K;

    v8f acc[2][4] = {};
    const int r = lane & 15, h = (lane >> 4) & 1;

    for (int kc = 0; kc < K; kc += 256) {
        __syncthreads();  // previous slice fully consumed
#if HAVE_TDM
        if (threadIdx.x == 0) {
            // low 32 bits of the flat shared address = LDS byte offset
            unsigned int lds_off = (unsigned int)(size_t)(void*)&sAT[0];
            tdm_load_a(lds_off, A + (size_t)m0 * K + kc, (unsigned int)K);
            __builtin_amdgcn_s_wait_tensorcnt(0);
        }
#else
        {   // fallback: cooperative staging, 128 threads x 64B, 2 rounds
            int t = threadIdx.x;
            for (int rep = 0; rep < 2; ++rep) {
                int idx  = t + rep * 128;            // 256 chunks of 32 elems
                int row  = idx >> 3, col = (idx & 7) * 32;
                const unsigned short* gp = A + (size_t)(m0 + row) * K + kc + col;
                uint4* dp = (uint4*)&sAT[row * LDA_P + col];
                dp[0] = ((const uint4*)gp)[0];
                dp[1] = ((const uint4*)gp)[1];
                dp[2] = ((const uint4*)gp)[2];
                dp[3] = ((const uint4*)gp)[3];
            }
        }
#endif
        __syncthreads();
#pragma unroll
        for (int k8 = 0; k8 < 256; k8 += 32) {
            v16bf a0 = load_a(sAT, LDA_P, k8, lane);
            v16bf a1 = load_a(sAT + 16 * LDA_P, LDA_P, k8, lane);
            const int k = kc + k8;
#pragma unroll
            for (int j = 0; j < 4; ++j) {
                v16bf bf = load_b(W0 + (size_t)(j * 16) * K, K, k, lane);
                acc[0][j] = wmma_bf16(a0, bf, acc[0][j]);
                acc[1][j] = wmma_bf16(a1, bf, acc[1][j]);
            }
        }
    }

#pragma unroll
    for (int i = 0; i < 2; ++i) {
#pragma unroll
        for (int j = 0; j < 4; ++j) {
            int   n  = n0 + j * 16 + r;
            float bn = bias ? bias[n] : 0.f;
#pragma unroll
            for (int v = 0; v < 8; ++v) {
                int   row = m0 + i * 16 + h * 8 + v;
                float val = acc[i][j][v] + bn;
                if constexpr (EPI == EPI_BF16) {
                    ((unsigned short*)out)[(size_t)row * N + n] = f32_to_bf16(val);
                } else if constexpr (EPI == EPI_GELU) {
                    float ge = 0.5f * val * (1.f + erff(val * 0.70710678118654752f));
                    ((unsigned short*)out)[(size_t)row * N + n] = f32_to_bf16(ge);
                } else if constexpr (EPI == EPI_PROJ) {
                    // fused window-reverse + roll(+4,+4) + residual -> x2
                    int    T = win_to_tok(row);
                    size_t o = (size_t)T * 256 + n;
                    x2out[o] = extra[o] + val;
                } else {  // EPI_FC2: d_out = x2 + mlp_out
                    size_t o           = (size_t)row * N + n;
                    ((float*)out)[o]   = extra[o] + val;
                }
            }
        }
    }
}

// ---------------------------------------------------------------- attention
// one wave per (window, head): QK^T (16 WMMA) -> LDS -> conv PE + RPB + mask
// + softmax (fp32, unnormalized bf16 probs in LDS) -> PV (16 WMMA).
__global__ __launch_bounds__(64) void k_attn(
    const unsigned short* __restrict__ qkv, const float* __restrict__ rpb,
    const float* __restrict__ pe_w, const float* __restrict__ pe_b,
    unsigned short* __restrict__ outw) {
    __shared__ float          sA[2][64 * 65];    // logits, fp32
    __shared__ unsigned short sP[2][64 * 72];    // probs, bf16 (16B-aligned rows)
    __shared__ float          sInv[2][64];       // 1/rowsum

    const int lane = threadIdx.x & 31;
    const int wid  = threadIdx.x >> 5;
    const int gh   = blockIdx.x * 2 + wid;
    const int win  = gh >> 3, head = gh & 7;

    const unsigned short* qb  = qkv + (size_t)win * 64 * 768 + head * 32;
    const unsigned short* kbp = qb + 256;
    const unsigned short* vbp = qb + 512;

    const int r = lane & 15, h = (lane >> 4) & 1;

    // ---- QK^T : q(64x32) * k(64x32)^T, K = HD = 32 (single WMMA per tile)
    v16bf aq[4], bk[4];
#pragma unroll
    for (int t = 0; t < 4; ++t) {
        aq[t] = load_a(qb + (size_t)(t * 16) * 768, 768, 0, lane);
        bk[t] = load_b(kbp + (size_t)(t * 16) * 768, 768, 0, lane);
    }
    v8f acc[4][4] = {};
#pragma unroll
    for (int ti = 0; ti < 4; ++ti)
#pragma unroll
        for (int tj = 0; tj < 4; ++tj)
            acc[ti][tj] = wmma_bf16(aq[ti], bk[tj], acc[ti][tj]);

    float* S = sA[wid];
#pragma unroll
    for (int ti = 0; ti < 4; ++ti)
#pragma unroll
        for (int tj = 0; tj < 4; ++tj)
#pragma unroll
            for (int v = 0; v < 8; ++v) {
                int i = ti * 16 + h * 8 + v, j = tj * 16 + r;
                S[i * 65 + j] = acc[ti][tj][v] * 0.17677669529663687f;  // 32^-0.5
            }

    // ---- conv PE + bias + RPB + shift mask + softmax (2 rows per lane)
    unsigned short* P = sP[wid];
    float wt[15];
#pragma unroll
    for (int t = 0; t < 15; ++t) wt[t] = pe_w[head * 15 + t];
    const float pb   = pe_b[head];
    const int   widx = win & 255, wh = widx >> 4, ww = widx & 15;

    for (int rr = 0; rr < 2; ++rr) {
        int   i = lane + rr * 32;
        int   ri = i >> 3, ci = i & 7;
        int   hi = wh * 8 + ri, wi = ww * 8 + ci;
        int   regi = (hi < 120 ? 0 : (hi < 124 ? 1 : 2)) * 3 +
                     (wi < 120 ? 0 : (wi < 124 ? 1 : 2));
        float m = -1e30f;
        for (int j = 0; j < 64; ++j) {
            float sv = S[i * 65 + j];
#pragma unroll
            for (int t = 0; t < 15; ++t) {
                int ii = i + t - 7;
                if (ii >= 0 && ii < 64) sv += wt[t] * S[ii * 65 + j];
            }
            int rj = j >> 3, cj = j & 7;
            sv += pb + rpb[((ri - rj + 7) * 15 + (ci - cj + 7)) * 8 + head];
            int hj = wh * 8 + rj, wj = ww * 8 + cj;
            int regj = (hj < 120 ? 0 : (hj < 124 ? 1 : 2)) * 3 +
                       (wj < 120 ? 0 : (wj < 124 ? 1 : 2));
            sv += (regj == regi) ? 0.f : -100.f;
            m = fmaxf(m, sv);
            P[i * 72 + j] = f32_to_bf16(sv);
        }
        float sum = 0.f;
        for (int j = 0; j < 64; ++j) {
            float e = __expf(bf16_to_f32(P[i * 72 + j]) - m);
            sum += e;
            P[i * 72 + j] = f32_to_bf16(e);   // unnormalized prob
        }
        sInv[wid][i] = 1.f / sum;
    }

    // ---- PV : P(64x64) * V(64x32), A frags straight from LDS (ds_load_b128)
    v8f o[4][2] = {};
#pragma unroll
    for (int kk = 0; kk < 2; ++kk) {
        v16bf pa[4];
#pragma unroll
        for (int t = 0; t < 4; ++t)
            pa[t] = load_a(P + (size_t)(t * 16) * 72, 72, kk * 32, lane);
        v16bf vb2[2];
#pragma unroll
        for (int tj = 0; tj < 2; ++tj) vb2[tj] = load_vT(vbp, tj * 16, kk * 32, lane);
#pragma unroll
        for (int ti = 0; ti < 4; ++ti)
#pragma unroll
            for (int tj = 0; tj < 2; ++tj)
                o[ti][tj] = wmma_bf16(pa[ti], vb2[tj], o[ti][tj]);
    }
    unsigned short* op = outw + (size_t)win * 64 * 256 + head * 32;
#pragma unroll
    for (int ti = 0; ti < 4; ++ti)
#pragma unroll
        for (int tj = 0; tj < 2; ++tj)
#pragma unroll
            for (int v = 0; v < 8; ++v) {
                int i = ti * 16 + h * 8 + v, dd = tj * 16 + r;
                op[(size_t)i * 256 + dd] =
                    f32_to_bf16(o[ti][tj][v] * sInv[wid][i]);
            }
}

// ---------------------------------------------------------------- launch
extern "C" void kernel_launch(void* const* d_in, const int* in_sizes, int n_in,
                              void* d_out, int out_size, void* d_ws,
                              size_t ws_size, hipStream_t stream) {
    const float* x       = (const float*)d_in[0];
    const float* n1g     = (const float*)d_in[1];
    const float* n1b     = (const float*)d_in[2];
    const float* qkv_w   = (const float*)d_in[3];
    const float* qkv_b   = (const float*)d_in[4];
    const float* rpb_t   = (const float*)d_in[5];
    const float* pe_w    = (const float*)d_in[6];
    const float* pe_b    = (const float*)d_in[7];
    const float* proj_w  = (const float*)d_in[8];
    const float* proj_b  = (const float*)d_in[9];
    const float* n2g     = (const float*)d_in[10];
    const float* n2b     = (const float*)d_in[11];
    const float* fc1_w   = (const float*)d_in[12];
    const float* fc1_b   = (const float*)d_in[13];
    const float* fc2_w   = (const float*)d_in[14];
    const float* fc2_b   = (const float*)d_in[15];

    // workspace layout (bytes)
    const size_t SZ_ACT  = (size_t)131072 * 256 * 2;   //  64 MiB bf16 activations
    const size_t SZ_BIG  = (size_t)131072 * 1024 * 2;  // 256 MiB (qkv / mlp hidden)
    const size_t SZ_X2   = (size_t)131072 * 256 * 4;   // 128 MiB fp32 residual
    char* ws = (char*)d_ws;
    unsigned short* winA = (unsigned short*)(ws);                       // LN1 out / LN2 out
    unsigned short* qkvb = (unsigned short*)(ws + SZ_ACT);              // qkv / mlp hidden
    unsigned short* outw = (unsigned short*)(ws + SZ_ACT + SZ_BIG);     // attn out
    float*          x2   = (float*)(ws + 2 * SZ_ACT + SZ_BIG);
    unsigned short* wq   = (unsigned short*)(ws + 2 * SZ_ACT + SZ_BIG + SZ_X2);
    unsigned short* wp   = wq + 768 * 256;
    unsigned short* w1   = wp + 256 * 256;
    unsigned short* w2   = w1 + 1024 * 256;
    unsigned short* hid  = qkvb;   // reuse after attention
    unsigned short* yin  = winA;   // reuse after qkv gemm

    // weights -> bf16 (N,K) row-major, matches WMMA B operand
    k_cvt<<<(768 * 256 + 255) / 256, 256, 0, stream>>>(qkv_w, wq, 768 * 256);
    k_cvt<<<(256 * 256 + 255) / 256, 256, 0, stream>>>(proj_w, wp, 256 * 256);
    k_cvt<<<(1024 * 256 + 255) / 256, 256, 0, stream>>>(fc1_w, w1, 1024 * 256);
    k_cvt<<<(1024 * 256 + 255) / 256, 256, 0, stream>>>(fc2_w, w2, 1024 * 256);

    // LN1 + shift + window partition
    k_ln<true><<<32768, 128, 0, stream>>>(x, n1g, n1b, winA);
    // QKV projection (131072 x 256) * (768 x 256)^T
    k_gemm<EPI_BF16><<<dim3(4096, 3), 128, 0, stream>>>(
        winA, wq, qkv_b, qkvb, 131072, 768, 256, nullptr, nullptr);
    // windowed attention with conv-PE / RPB / shift mask
    k_attn<<<8192, 64, 0, stream>>>(qkvb, rpb_t, pe_w, pe_b, outw);
    // proj + window reverse + unshift + residual -> x2
    k_gemm<EPI_PROJ><<<dim3(4096, 1), 128, 0, stream>>>(
        outw, wp, proj_b, nullptr, 131072, 256, 256, x, x2);
    // LN2
    k_ln<false><<<32768, 128, 0, stream>>>(x2, n2g, n2b, yin);
    // MLP fc1 + exact GELU
    k_gemm<EPI_GELU><<<dim3(4096, 4), 128, 0, stream>>>(
        yin, w1, fc1_b, hid, 131072, 1024, 256, nullptr, nullptr);
    // MLP fc2 + residual -> d_out (fp32)
    k_gemm<EPI_FC2><<<dim3(4096, 1), 128, 0, stream>>>(
        hid, w2, fc2_b, d_out, 131072, 256, 1024, x2, nullptr);
}
